// GMM_43439299232234
// MI455X (gfx1250) — matrix operand order repaired and checked by reading, not compiled, not run on hardware
//
#include <hip/hip_runtime.h>
#include <hip/hip_bf16.h>
#include <math.h>

// GMM sufficient statistics + energies, MI455X (gfx1250, wave32, WMMA bf16).
//
// Pipeline (all on `stream`, deterministic, partials instead of float atomics):
//   stats_kernel    : P_k = sum_i gamma[i,k] z_i z_i^T  via v_wmma_f32_16x16x32_bf16,
//                     z chunks staged by the Tensor Data Mover (tensor_load_to_lds,
//                     double-buffered, s_wait_tensorcnt) when available
//   moments_kernel  : gamma_sum[k], sum_i gamma[i,k] z_i   (f32)
//   finalize_kernel : sigma = P/gs - mu mu^T + jitter, Cholesky, M = L^-1,
//                     b = M mu, c_k = log(phi)+64 log(2pi)+logdet,
//                     and M^T pre-swizzled into WMMA B-fragment layout (bf16)
//   energy_kernel   : q = ||M z - b||^2 via WMMA, online logsumexp -> energies

#define NS 65536
#define KC 16
#define DD 64
#define JITTER 1e-6f
#define LOG2PI 1.8378770664093453f

typedef __attribute__((ext_vector_type(16))) __bf16 bf16x16;
typedef __attribute__((ext_vector_type(8)))  float  f32x8;

union Frag {
    bf16x16 v;
    __bf16  e[16];
    uint4   q[2];
};

// Contiguous 16 x bf16 (32B) from LDS/global -> one fragment.
__device__ inline Frag frag_contig(const __bf16* p) {
    Frag f;
    f.q[0] = *(const uint4*)(p);
    f.q[1] = *(const uint4*)(p + 8);
    return f;
}

// A-fragment: two 8-element chunks at p and p+16 (16-bit A 16x32 layout:
// lane<16 holds K 0..7 & 16..23; lane>=16 holds K 8..15 & 24..31).
__device__ inline Frag frag_a(const __bf16* p) {
    Frag f;
    f.q[0] = *(const uint4*)(p);
    f.q[1] = *(const uint4*)(p + 16);
    return f;
}

// ---------------------------------------------------------------------------
// Tensor Data Mover support (gfx1250). Builtin exists on both toolchains but
// with different arity: 5 args (ROCm 7.2 / clang-22), 6 args (clang-23+).
// ---------------------------------------------------------------------------
#if defined(__HIP_DEVICE_COMPILE__) && defined(__has_builtin)
#  if __has_builtin(__builtin_amdgcn_tensor_load_to_lds)
#    define HAVE_TDM 1
#  endif
#endif
#ifndef HAVE_TDM
#  define HAVE_TDM 0
#endif

#if HAVE_TDM
typedef __attribute__((ext_vector_type(4))) unsigned int u32x4t;
typedef __attribute__((ext_vector_type(4))) int          i32x4t;
typedef __attribute__((ext_vector_type(8))) int          i32x8t;

// LDS byte offset of a __shared__ object (generic -> addrspace(3) -> int).
__device__ inline unsigned lds_byte_off(const void* p) {
    return (unsigned)(unsigned long long)
        (const __attribute__((address_space(3))) char*)p;
}

// Issue a 2-D TDM load: 64 rows x 64 f32 (one z chunk), row stride 64 elems.
// D# packing per CDNA5 ISA 8.3/8.4: group0 = {count=1, lds_addr, global_addr,
// type=2}; group1 = {data_size=4B, tensor_dim0=64, tensor_dim1=65536,
// tile_dim0=64, tile_dim1=64, tensor_dim0_stride=64}.
__device__ inline void tdm_load_z_chunk(unsigned ldsOff, const float* gsrc) {
    const unsigned long long ga = (unsigned long long)gsrc;
    u32x4t g0;
    g0[0] = 1u;                                            // count=1, user mode
    g0[1] = ldsOff;                                        // lds_addr (bytes)
    g0[2] = (unsigned)ga;                                  // global_addr[31:0]
    g0[3] = (unsigned)((ga >> 32) & 0x1ffffffull)          // global_addr[56:32]
          | (2u << 30);                                    // type=2 ("image")
    i32x8t g1;
    g1[0] = (int)(2u << 16);                               // data_size=2 (4B)
    g1[1] = (int)(64u << 16);                              // tensor_dim0=64 [15:0]
    g1[2] = 0;                                             // dim0 hi, dim1 lo
    g1[3] = (int)((64u << 16) | 1u);                       // tile_dim0=64 | dim1 hi (65536)
    g1[4] = 64;                                            // tile_dim1=64
    g1[5] = 64;                                            // tensor_dim0_stride=64
    g1[6] = 0;
    g1[7] = 0;
    i32x4t z4; z4[0] = 0; z4[1] = 0; z4[2] = 0; z4[3] = 0; // 2-D: groups 2/3 unused
#if __clang_major__ >= 23
    i32x8t z8;
#pragma unroll
    for (int i = 0; i < 8; ++i) z8[i] = 0;
    __builtin_amdgcn_tensor_load_to_lds(g0, g1, z4, z4, z8, 0);
#else
    __builtin_amdgcn_tensor_load_to_lds(g0, g1, z4, z4, 0);
#endif
}
#endif  // HAVE_TDM

// ---------------------------------------------------------------------------
// Kernel 1: weighted second moments P_k, partial per block slice.
// Grid: 64 blocks x 512 threads (16 waves). Wave w owns component k = w.
// Each block processes 1024 samples in chunks of 64. z chunks arrive via
// double-buffered TDM DMA (f32), then are repacked to feature-major bf16 in
// LDS, which serves both the A and B per-lane WMMA fragment layouts.
// ---------------------------------------------------------------------------
__global__ void __launch_bounds__(512)
stats_kernel(const float* __restrict__ z, const float* __restrict__ gamma,
             float* __restrict__ Spart) {
#if HAVE_TDM
    __shared__ __align__(16) float  zRaw[2][64 * DD];  // 2 x 16KB TDM destinations
#endif
    __shared__ __align__(16) __bf16 zA[DD * 64];       // [feature][sample], 8KB
    __shared__ __align__(16) __bf16 wT[KC * 64];       // [k][sample], 2KB

    const int tid  = threadIdx.x;
    const int lane = tid & 31;
    const int k    = tid >> 5;            // wave id == mixture component
    const int hi16 = lane >> 4;           // 0: lanes 0-15, 1: lanes 16-31
    const int lo16 = lane & 15;
    const int n0   = blockIdx.x * 1024;

    f32x8 acc[4][4];
#pragma unroll
    for (int mt = 0; mt < 4; ++mt)
#pragma unroll
        for (int nt = 0; nt < 4; ++nt)
#pragma unroll
            for (int j = 0; j < 8; ++j) acc[mt][nt][j] = 0.0f;

#if HAVE_TDM
    if (tid < 32)  // wave 0 drives the DMA pipeline
        tdm_load_z_chunk(lds_byte_off(zRaw[0]), z + (size_t)n0 * DD);
#endif

    for (int c0 = 0; c0 < 1024; c0 += 64) {
#if HAVE_TDM
        const int buf = (c0 >> 6) & 1;
        if (tid < 32) {
            __builtin_amdgcn_s_wait_tensorcnt(0);      // chunk c0 landed in zRaw[buf]
            if (c0 + 64 < 1024)                        // prefetch next chunk via TDM
                tdm_load_z_chunk(lds_byte_off(zRaw[buf ^ 1]),
                                 z + (size_t)(n0 + c0 + 64) * DD);
        }
        __syncthreads();
        // ---- repack f32 chunk -> feature-major bf16 ----
        {
            const int s  = tid >> 3;
            const int d0 = (tid & 7) * 8;
            const float* src = &zRaw[buf][s * DD + d0];
#pragma unroll
            for (int j = 0; j < 8; ++j)
                zA[(d0 + j) * 64 + s] = (__bf16)src[j];
        }
#else
        // ---- synchronous staging fallback ----
        {
            const int s  = tid >> 3;
            const int d0 = (tid & 7) * 8;
            const float* src = z + (size_t)(n0 + c0 + s) * DD + d0;
#pragma unroll
            for (int j = 0; j < 8; ++j)
                zA[(d0 + j) * 64 + s] = (__bf16)src[j];
        }
        if (c0 + 64 < 1024)
            __builtin_prefetch(z + (size_t)(n0 + c0 + 64 + (tid >> 3)) * DD, 0, 1);
#endif
        // ---- stage gamma chunk, [k][sample] bf16 ----
#pragma unroll
        for (int i = 0; i < 2; ++i) {
            const int e  = tid + i * 512;
            const int s  = e >> 4;
            const int kk = e & 15;
            wT[kk * 64 + s] = (__bf16)gamma[(size_t)(n0 + c0 + s) * KC + kk];
        }
        __syncthreads();

#pragma unroll
        for (int s0 = 0; s0 < 64; s0 += 32) {
            // per-lane weight vector w_k[s0 + 16*hi16 .. +15]
            Frag wf = frag_contig(&wT[k * 64 + s0 + 16 * hi16]);

            Frag af[4];
#pragma unroll
            for (int mt = 0; mt < 4; ++mt)
                af[mt] = frag_contig(&zA[(16 * mt + lo16) * 64 + s0 + 16 * hi16]);

            Frag bf[4];
#pragma unroll
            for (int nt = 0; nt < 4; ++nt) {
                Frag zb = frag_contig(&zA[(16 * nt + lo16) * 64 + s0 + 16 * hi16]);
#pragma unroll
                for (int i = 0; i < 16; ++i)
                    bf[nt].e[i] = (__bf16)((float)zb.e[i] * (float)wf.e[i]);
            }

#pragma unroll
            for (int mt = 0; mt < 4; ++mt)
#pragma unroll
                for (int nt = 0; nt < 4; ++nt)
                    acc[mt][nt] = __builtin_amdgcn_wmma_f32_16x16x32_bf16(
                        false, af[mt].v, false, bf[nt].v,
                        (short)0, acc[mt][nt], false, false);
        }
        __syncthreads();
    }

    // write partial P_k for this slice; C layout: VGPR j, lane l ->
    // m = 16*mt + j + 8*(l>=16), n = 16*nt + (l&15)
    float* dst = Spart + ((size_t)blockIdx.x * KC + k) * (DD * DD);
#pragma unroll
    for (int mt = 0; mt < 4; ++mt)
#pragma unroll
        for (int nt = 0; nt < 4; ++nt)
#pragma unroll
            for (int j = 0; j < 8; ++j)
                dst[(16 * mt + j + 8 * hi16) * DD + 16 * nt + lo16] = acc[mt][nt][j];
}

// ---------------------------------------------------------------------------
// Kernel 2: gamma_sum and mu numerator (f32, partial per block).
// Grid: 64 blocks x 1024 threads; thread (k,d) accumulates over 1024 samples.
// ---------------------------------------------------------------------------
__global__ void __launch_bounds__(1024)
moments_kernel(const float* __restrict__ z, const float* __restrict__ gamma,
               float* __restrict__ gpart, float* __restrict__ mpart) {
    __shared__ float sz[64 * DD];   // 16KB
    __shared__ float sg[64 * KC];   // 4KB
    const int tid = threadIdx.x;
    const int k = tid >> 6, d = tid & 63;
    const int n0 = blockIdx.x * 1024;

    float acc = 0.0f, gacc = 0.0f;
    for (int c0 = 0; c0 < 1024; c0 += 64) {
#pragma unroll
        for (int i = 0; i < 4; ++i)
            sz[tid + i * 1024] = z[(size_t)(n0 + c0) * DD + tid + i * 1024];
        sg[tid] = gamma[(size_t)(n0 + c0) * KC + tid];
        __syncthreads();
        for (int s = 0; s < 64; ++s) {
            const float g = sg[s * KC + k];
            acc  += g * sz[s * DD + d];
            gacc += g;
        }
        __syncthreads();
    }
    mpart[((size_t)blockIdx.x * KC + k) * DD + d] = acc;
    if (d == 0) gpart[blockIdx.x * KC + k] = gacc;
}

// ---------------------------------------------------------------------------
// Kernel 3: per-component finalize. Grid: 16 blocks x 256 threads.
// ---------------------------------------------------------------------------
__global__ void __launch_bounds__(256)
finalize_kernel(const float* __restrict__ Spart, const float* __restrict__ gpart,
                const float* __restrict__ mpart,
                __bf16* __restrict__ Mb, float* __restrict__ bvec,
                float* __restrict__ cconst) {
    __shared__ float sA[DD][DD + 1];   // sigma -> L (lower), padded vs bank conflicts
    __shared__ float sM[DD][DD + 1];   // L^-1
    __shared__ float sMu[DD];
    __shared__ float sGs;
    const int tid = threadIdx.x;
    const int k   = blockIdx.x;

    if (tid == 0) {
        float g = 0.0f;
        for (int s = 0; s < 64; ++s) g += gpart[s * KC + k];
        sGs = g;
    }
    __syncthreads();
    const float invgs = 1.0f / sGs;

    if (tid < DD) {
        float m = 0.0f;
        for (int s = 0; s < 64; ++s) m += mpart[((size_t)s * KC + k) * DD + tid];
        sMu[tid] = m * invgs;
    }
    __syncthreads();

    // sigma = (sum slices P)/gs - mu mu^T + jitter I
    for (int e = tid; e < DD * DD; e += 256) {
        const int m = e >> 6, n = e & 63;
        float s = 0.0f;
        for (int sl = 0; sl < 64; ++sl)
            s += Spart[(((size_t)sl * KC + k) << 12) + e];
        sA[m][n] = s * invgs - sMu[m] * sMu[n] + (m == n ? JITTER : 0.0f);
    }
    __syncthreads();

    // In-LDS right-looking Cholesky (lower triangle)
    for (int j = 0; j < DD; ++j) {
        if (tid == 0) sA[j][j] = sqrtf(sA[j][j]);
        __syncthreads();
        const float inv = 1.0f / sA[j][j];
        const int i = j + 1 + tid;
        if (i < DD) sA[i][j] *= inv;
        __syncthreads();
        if (i < DD) {
            const float lij = sA[i][j];
            for (int p = j + 1; p <= i; ++p) sA[i][p] -= lij * sA[p][j];
        }
        __syncthreads();
    }

    if (tid == 0) {
        float ld = 0.0f;
        for (int j = 0; j < DD; ++j) ld += logf(sA[j][j]);
        // logits = log(phi) - 0.5 q + d*log(2pi) + logdet  (matches reference signs)
        cconst[k] = logf(sGs * (1.0f / (float)NS)) + 64.0f * LOG2PI + 2.0f * ld;
    }

    // M = L^-1 (lower triangular), one column per thread
    if (tid < DD) {
        const int c = tid;
        for (int i = 0; i < c; ++i) sM[i][c] = 0.0f;
        sM[c][c] = 1.0f / sA[c][c];
        for (int i = c + 1; i < DD; ++i) {
            float s = 0.0f;
            for (int p = c; p < i; ++p) s += sA[i][p] * sM[p][c];
            sM[i][c] = -s / sA[i][i];
        }
    }
    __syncthreads();

    // b = M mu
    if (tid < DD) {
        float s = 0.0f;
        for (int c = 0; c < DD; ++c) s += sM[tid][c] * sMu[c];
        bvec[k * DD + tid] = s;
    }

    // Pre-swizzle M^T into this code's WMMA B-fragment per-lane layout (contiguous k order, same as A) (bf16):
    // fragment (k, t, nt), lane l, elem p  ->  B[f][r] = M[r][f],
    // r = 16*nt + (l&15), f = 32*t + 16*(l>>4) + p.
    for (int idx = tid; idx < 2 * 4 * 32 * 16; idx += 256) {
        const int p  = idx & 15;
        const int ln = (idx >> 4) & 31;
        const int nt = (idx >> 9) & 3;
        const int t  = idx >> 11;
        const int r  = nt * 16 + (ln & 15);
        const int f  = t * 32 + 16 * (ln >> 4) + p;
        Mb[(size_t)k * 4096 + idx] = (__bf16)sM[r][f];
    }
}

// ---------------------------------------------------------------------------
// Kernel 4: energies. Grid: 512 blocks x 256 threads (8 waves, 16 samples/wave).
// Dynamic LDS: all 16 swizzled M matrices (128KB) + b (4KB) + c (64B).
// ---------------------------------------------------------------------------
__global__ void __launch_bounds__(256)
energy_kernel(const float* __restrict__ z, const __bf16* __restrict__ Mb,
              const float* __restrict__ bvec, const float* __restrict__ cconst,
              float* __restrict__ out) {
    extern __shared__ __align__(16) unsigned char smem[];
    __bf16* sMb = (__bf16*)smem;                               // 128KB
    float*  sB  = (float*)(smem + (size_t)KC * 4096 * 2);      // 4KB
    float*  sC  = sB + KC * DD;                                // 64B

    const int tid  = threadIdx.x;
    const int lane = tid & 31;
    const int wid  = tid >> 5;
    const int hi16 = lane >> 4;
    const int lo16 = lane & 15;

    // stage swizzled M fragments + constants
    for (int e = tid; e < KC * 4096 / 8; e += 256)
        ((uint4*)sMb)[e] = ((const uint4*)Mb)[e];
    for (int e = tid; e < KC * DD; e += 256) sB[e] = bvec[e];
    if (tid < KC) sC[tid] = cconst[tid];
    __syncthreads();

    const int i0   = (blockIdx.x * 8 + wid) * 16;
    const int srow = i0 + lo16;

    // A fragments: z rows (samples), bf16, contiguous K order (k = 16*hi16 + elem)
    Frag af[2];
#pragma unroll
    for (int t = 0; t < 2; ++t) {
        const float* p = z + (size_t)srow * DD + 32 * t + 16 * hi16;
#pragma unroll
        for (int j = 0; j < 8; ++j) af[t].e[j]     = (__bf16)p[j];
#pragma unroll
        for (int j = 0; j < 8; ++j) af[t].e[8 + j] = (__bf16)p[8 + j];
    }

    float m8[8], s8[8];
#pragma unroll
    for (int j = 0; j < 8; ++j) { m8[j] = -1e30f; s8[j] = 0.0f; }

    for (int k = 0; k < KC; ++k) {
        f32x8 accv[4];
#pragma unroll
        for (int nt = 0; nt < 4; ++nt)
#pragma unroll
            for (int j = 0; j < 8; ++j) accv[nt][j] = 0.0f;

#pragma unroll
        for (int t = 0; t < 2; ++t)
#pragma unroll
            for (int nt = 0; nt < 4; ++nt) {
                Frag bfr = frag_contig(&sMb[(size_t)((k * 2 + t) * 4 + nt) * 512 + lane * 16]);
                accv[nt] = __builtin_amdgcn_wmma_f32_16x16x32_bf16(
                    false, af[t].v, false, bfr.v, (short)0, accv[nt], false, false);
            }

        // q_i = || y_i - b ||^2 ; reduce over r (lane&15 dimension)
        float q[8];
#pragma unroll
        for (int j = 0; j < 8; ++j) q[j] = 0.0f;
#pragma unroll
        for (int nt = 0; nt < 4; ++nt) {
            const float bv = sB[k * DD + 16 * nt + lo16];
#pragma unroll
            for (int j = 0; j < 8; ++j) {
                const float d = accv[nt][j] - bv;
                q[j] += d * d;
            }
        }
#pragma unroll
        for (int off = 1; off < 16; off <<= 1)
#pragma unroll
            for (int j = 0; j < 8; ++j) q[j] += __shfl_xor(q[j], off, 32);

        const float cc = sC[k];
#pragma unroll
        for (int j = 0; j < 8; ++j) {
            const float logit = cc - 0.5f * q[j];
            const float mn = fmaxf(m8[j], logit);
            s8[j] = s8[j] * expf(m8[j] - mn) + expf(logit - mn);
            m8[j] = mn;
        }
    }

    if (lo16 == 0) {
        const int base = i0 + 8 * hi16;
#pragma unroll
        for (int j = 0; j < 8; ++j)
            out[base + j] = -(m8[j] + logf(s8[j]));
    }
}

// ---------------------------------------------------------------------------
extern "C" void kernel_launch(void* const* d_in, const int* in_sizes, int n_in,
                              void* d_out, int out_size, void* d_ws, size_t ws_size,
                              hipStream_t stream) {
    (void)in_sizes; (void)n_in; (void)out_size; (void)ws_size;
    const float* z     = (const float*)d_in[0];
    const float* gamma = (const float*)d_in[1];
    float*       out   = (float*)d_out;

    unsigned char* ws = (unsigned char*)d_ws;
    size_t off = 0;
    float* Spart = (float*)(ws + off); off += (size_t)64 * KC * DD * DD * 4; // 16 MB
    float* gpart = (float*)(ws + off); off += (size_t)64 * KC * 4;
    float* mpart = (float*)(ws + off); off += (size_t)64 * KC * DD * 4;
    __bf16* Mb   = (__bf16*)(ws + off); off += (size_t)KC * 4096 * 2;        // 128 KB
    float* bvec  = (float*)(ws + off); off += (size_t)KC * DD * 4;
    float* cconst= (float*)(ws + off); off += (size_t)KC * 4;

    stats_kernel  <<<64, 512, 0, stream>>>(z, gamma, Spart);
    moments_kernel<<<64, 1024, 0, stream>>>(z, gamma, gpart, mpart);
    finalize_kernel<<<KC, 256, 0, stream>>>(Spart, gpart, mpart, Mb, bvec, cconst);

    const size_t dynLds = (size_t)KC * 4096 * 2 + (size_t)KC * DD * 4 + KC * 4;
    energy_kernel <<<NS / 128, 256, dynLds, stream>>>(z, Mb, bvec, cconst, out);
}
